// MambaBlock_51127290692383
// MI455X (gfx1250) — matrix operand order repaired and checked by reading, not compiled, not run on hardware
//
#include <hip/hip_runtime.h>
#include <hip/hip_bf16.h>
#include <stdint.h>

typedef __bf16 bf16_t;
typedef __attribute__((ext_vector_type(16))) __bf16 v16bf;
typedef __attribute__((ext_vector_type(8)))  __bf16 v8bf;
typedef __attribute__((ext_vector_type(8)))  float  v8f;

#define B_SZ     2
#define L_SZ     2048
#define D_MODEL  1024
#define D_INNER  2048
#define D_STATE  16
#define D_CONV   4
#define DT_RANK  128
#define M_ROWS   (B_SZ * L_SZ)           // 4096
#define N_XZ     (2 * D_INNER)           // 4096
#define N_XPROJ  (DT_RANK + 2 * D_STATE) // 160 (logical)
#define N_XPP    256                     // padded to multiple of BN for guard-free GEMM

#define BM 128
#define BN 128
#define BK 32
#define BKP 40                            // padded Bs row (elems): 80B stride, no bank conflicts
#define SMEM_A (2 * BM * BK * 2)          // 16384 B
#define SMEM_B (2 * BN * BKP * 2)         // 20480 B
#define SMEM_BYTES (SMEM_A + SMEM_B)

// ---------------------------------------------------------------------------
// Tensor Data Mover (gfx1250): async 2D tile DMA global -> LDS.
// D# per CDNA5 ISA ch.8: group0 {count, lds_addr, global_addr, type=2},
// group1 {data_size=2B, tensor_dim0/1, tile_dim0/1, tensor_dim0_stride}.
// ---------------------------------------------------------------------------
#if defined(__has_builtin)
#if __has_builtin(__builtin_amdgcn_tensor_load_to_lds)
#define HAVE_TDM 1
#endif
#endif

#ifdef HAVE_TDM
typedef __attribute__((ext_vector_type(4))) unsigned int v4u;
typedef __attribute__((ext_vector_type(8))) int          v8i;
typedef __attribute__((ext_vector_type(4))) int          v4i;

__device__ __forceinline__ void tdm_load_tile(const bf16_t* gaddr,
                                              unsigned lds_byte_off,
                                              int K, int M) {
  const unsigned long long ga = (unsigned long long)(uintptr_t)gaddr;
  v4u g0;
  g0[0] = 1u;                                        // count=1, user mode
  g0[1] = lds_byte_off;                              // lds_addr [63:32]
  g0[2] = (unsigned)(ga & 0xFFFFFFFFu);              // global_addr lo
  g0[3] = (unsigned)((ga >> 32) & 0x1FFFFFFu)        // global_addr hi (57b)
        | (2u << 30);                                // type=2 ("image")
  v8i g1;
  g1[0] = 0x00010000;                                // data_size=1 (2B), mask=0
  g1[1] = (int)(((unsigned)K & 0xFFFFu) << 16);      // tensor_dim0 lo16 @63:48
  g1[2] = (int)((((unsigned)K >> 16) & 0xFFFFu)      // tensor_dim0 hi16 @79:64
        |       (((unsigned)M & 0xFFFFu) << 16));    // tensor_dim1 lo16 @95:80
  g1[3] = (int)((((unsigned)M >> 16) & 0xFFFFu)      // tensor_dim1 hi16 @111:96
        |       ((unsigned)BK << 16));               // tile_dim0 @127:112
  g1[4] = BM;                                        // tile_dim1 @143:128
  g1[5] = (int)(unsigned)K;                          // dim0_stride lo32 @191:160
  g1[6] = 0;                                         // dim0_stride hi / dim1_stride
  g1[7] = 0;
  const v4i z4 = {0, 0, 0, 0};
#if defined(__clang_major__) && (__clang_major__ >= 23)
  const v8i z8 = {0, 0, 0, 0, 0, 0, 0, 0};
  __builtin_amdgcn_tensor_load_to_lds(g0, g1, z4, z4, z8, 0);
#else
  __builtin_amdgcn_tensor_load_to_lds(g0, g1, z4, z4, 0);
#endif
}
#endif  // HAVE_TDM

// ---------------------------------------------------------------------------
// f32 -> bf16 conversion (grid-stride)
// ---------------------------------------------------------------------------
__global__ void cvt_f32_bf16_kernel(const float* __restrict__ src,
                                    bf16_t* __restrict__ dst, int n) {
  for (int i = blockIdx.x * blockDim.x + threadIdx.x; i < n;
       i += gridDim.x * blockDim.x)
    dst[i] = (bf16_t)src[i];
}

// W_xproj [D_INNER][160] f32 -> [D_INNER][256] bf16, zero-padded cols 160..255
__global__ void cvt_pad_xproj_kernel(const float* __restrict__ src,
                                     bf16_t* __restrict__ dst) {
  const int total = D_INNER * N_XPP;
  for (int i = blockIdx.x * blockDim.x + threadIdx.x; i < total;
       i += gridDim.x * blockDim.x) {
    const int n = i & (N_XPP - 1);
    const int k = i >> 8;
    dst[i] = (n < N_XPROJ) ? (bf16_t)src[k * N_XPROJ + n] : (bf16_t)0.0f;
  }
}

// ---------------------------------------------------------------------------
// LayerNorm: one block per token row, writes bf16 activations for GEMM-A
// ---------------------------------------------------------------------------
__global__ __launch_bounds__(256)
void layernorm_kernel(const float* __restrict__ x,
                      const float* __restrict__ gamma,
                      const float* __restrict__ beta,
                      bf16_t* __restrict__ xn) {
  const int row = blockIdx.x;
  const int tid = threadIdx.x;
  const float* xr = x + (size_t)row * D_MODEL;
  float s = 0.f, s2 = 0.f;
  for (int i = tid; i < D_MODEL; i += 256) {
    float v = xr[i];
    s += v; s2 += v * v;
  }
  __shared__ float red0[256], red1[256];
  red0[tid] = s; red1[tid] = s2;
  __syncthreads();
  for (int off = 128; off > 0; off >>= 1) {
    if (tid < off) { red0[tid] += red0[tid + off]; red1[tid] += red1[tid + off]; }
    __syncthreads();
  }
  const float mu  = red0[0] * (1.0f / D_MODEL);
  const float var = red1[0] * (1.0f / D_MODEL) - mu * mu;
  const float rstd = rsqrtf(var + 1e-5f);
  bf16_t* xo = xn + (size_t)row * D_MODEL;
  for (int i = tid; i < D_MODEL; i += 256)
    xo[i] = (bf16_t)((xr[i] - mu) * rstd * gamma[i] + beta[i]);
}

// ---------------------------------------------------------------------------
// Guard-free WMMA bf16 GEMM: C[M,N] = A[M,K] * B[K,N] (+bias)(+act)(+residual)
// Requires M % 128 == 0, N % 128 == 0, K % 32 == 0.
// A tiles: TDM async DMA (tensor_load_to_lds + s_wait_tensorcnt), double-buffered.
// B tiles: global->reg->transposed LDS store, overlapped with WMMAs.
// ACT: 0 = none, 1 = softplus
// ---------------------------------------------------------------------------
template <int ACT>
__global__ __launch_bounds__(256)
void wmma_gemm_kernel(const bf16_t* __restrict__ A,
                      const bf16_t* __restrict__ Bg,
                      float* __restrict__ C,
                      const float* __restrict__ bias,
                      const float* __restrict__ residual,
                      bf16_t* __restrict__ Cbf,
                      int M, int N, int K, int bf_cols) {
  extern __shared__ char smem[];  // dynamic LDS -> descriptor offsets start at 0
  bf16_t* const AsBase = (bf16_t*)smem;               // [2][BM][BK]
  bf16_t* const BsBase = (bf16_t*)(smem + SMEM_A);    // [2][BN][BKP]

  const int tid   = threadIdx.x;
  const int lane  = tid & 31;
  const int wid   = tid >> 5;
  const int waveM = (wid & 1) * 64;   // 2 wave rows
  const int waveN = (wid >> 1) * 32;  // 4 wave cols
  const int m0 = blockIdx.y * BM;
  const int n0 = blockIdx.x * BN;
  const int r    = lane & 15;
  const int half = lane >> 4;

  v8f acc[4][2];
#pragma unroll
  for (int mt = 0; mt < 4; ++mt)
#pragma unroll
    for (int nt = 0; nt < 2; ++nt)
#pragma unroll
      for (int v = 0; v < 8; ++v) acc[mt][nt][v] = 0.0f;

#ifndef HAVE_TDM
  v8bf aReg[2];
#endif
  v8bf bReg[2];

  auto load_a = [&](int k0, int buf) {
#ifdef HAVE_TDM
    if (tid == 0)  // TDM ignores EXEC; single issue per workgroup (wave 0)
      tdm_load_tile(A + (size_t)m0 * K + k0,
                    (unsigned)(buf * BM * BK * 2), K, M);
#else
#pragma unroll
    for (int i = 0; i < 2; ++i) {
      const int c = tid + i * 256;
      aReg[i] = *(const v8bf*)(A + (size_t)(m0 + (c >> 2)) * K + k0 + (c & 3) * 8);
    }
    (void)buf;
#endif
  };
  auto store_a = [&](int buf) {
#ifndef HAVE_TDM
#pragma unroll
    for (int i = 0; i < 2; ++i) {
      const int c = tid + i * 256;
      *(v8bf*)(AsBase + buf * BM * BK + (c >> 2) * BK + (c & 3) * 8) = aReg[i];
    }
#else
    (void)buf;
#endif
  };
  auto wait_a = [&]() {
#ifdef HAVE_TDM
    if (tid == 0) __builtin_amdgcn_s_wait_tensorcnt(0);
#endif
  };
  auto load_b = [&](int k0) {
#pragma unroll
    for (int i = 0; i < 2; ++i) {
      const int c = tid + i * 256;  // 0..511: 32 K-rows x 16 col-chunks
      bReg[i] = *(const v8bf*)(Bg + (size_t)(k0 + (c >> 4)) * N + n0 + (c & 15) * 8);
    }
  };
  auto store_b = [&](int buf) {   // transposed: Bs[n][k], padded row BKP
#pragma unroll
    for (int i = 0; i < 2; ++i) {
      const int c = tid + i * 256;
#pragma unroll
      for (int j = 0; j < 8; ++j)
        BsBase[buf * BN * BKP + ((c & 15) * 8 + j) * BKP + (c >> 4)] = bReg[i][j];
    }
  };

  const int KT = K / BK;
  load_a(0, 0);
  load_b(0);
  store_a(0);
  store_b(0);
  wait_a();
  __syncthreads();

  for (int kt = 0; kt < KT; ++kt) {
    const int buf = kt & 1;
    if (kt + 1 < KT) {
      load_a((kt + 1) * BK, buf ^ 1);  // TDM DMA flies under the WMMAs
      load_b((kt + 1) * BK);           // global loads fly under the WMMAs
    }
    if (kt + 2 < KT) {                 // warm GL2 (global_prefetch_b8)
      const int k2 = (kt + 2) * BK;
      __builtin_prefetch(Bg + (size_t)(k2 + (tid >> 4)) * N + n0 + (tid & 15) * 8, 0, 0);
    }

    // ---- A fragments: elems 0..7 <-> K=half*8+j, elems 8..15 <-> K=16+half*8+j
    const bf16_t* const as = AsBase + buf * BM * BK;
    v16bf afrag[4];
#pragma unroll
    for (int mt = 0; mt < 4; ++mt) {
      const bf16_t* p = as + (waveM + mt * 16 + r) * BK + half * 8;
      v8bf lo = *(const v8bf*)p;
      v8bf hi = *(const v8bf*)(p + 16);
#pragma unroll
      for (int j = 0; j < 8; ++j) { afrag[mt][j] = lo[j]; afrag[mt][j + 8] = hi[j]; }
    }
    // ---- B fragments: lane holds column, elems 0..7 <-> K=half*8+j, elems 8..15 <-> K=16+half*8+j
    const bf16_t* const bs = BsBase + buf * BN * BKP;
    v16bf bfrag[2];
#pragma unroll
    for (int nt = 0; nt < 2; ++nt) {
      const bf16_t* p = bs + (waveN + nt * 16 + r) * BKP + half * 8;
      v8bf lo = *(const v8bf*)p;
      v8bf hi = *(const v8bf*)(p + 16);
#pragma unroll
      for (int j = 0; j < 8; ++j) { bfrag[nt][j] = lo[j]; bfrag[nt][j + 8] = hi[j]; }
    }
#pragma unroll
    for (int mt = 0; mt < 4; ++mt)
#pragma unroll
      for (int nt = 0; nt < 2; ++nt)
        acc[mt][nt] = __builtin_amdgcn_wmma_f32_16x16x32_bf16(
            false, afrag[mt], false, bfrag[nt], (short)0, acc[mt][nt],
            false, false);

    if (kt + 1 < KT) {
      store_a(buf ^ 1);
      store_b(buf ^ 1);
      wait_a();                 // only wave 0 stalls on TENSORcnt, post-compute
    }
    __syncthreads();
  }

  // ---- epilogue: C/D layout: row = base + half*8 + v, col = base + r
#pragma unroll
  for (int mt = 0; mt < 4; ++mt)
#pragma unroll
    for (int nt = 0; nt < 2; ++nt) {
      const int col = n0 + waveN + nt * 16 + r;
#pragma unroll
      for (int v = 0; v < 8; ++v) {
        const int row = m0 + waveM + mt * 16 + half * 8 + v;
        float val = acc[mt][nt][v];
        if (bias) val += bias[col];
        if (ACT == 1)  // softplus
          val = (val > 20.f) ? val : __logf(1.f + __expf(val));
        if (residual) val += residual[(size_t)row * N + col];
        C[(size_t)row * N + col] = val;
        if (Cbf && col < bf_cols)
          Cbf[(size_t)row * bf_cols + col] = (bf16_t)val;
      }
    }
}

// ---------------------------------------------------------------------------
// Depthwise causal conv (taps=4) + bias + SiLU -> u (f32 for scan, bf16 for GEMM)
// ---------------------------------------------------------------------------
__global__ void conv_silu_kernel(const float* __restrict__ xz,
                                 const float* __restrict__ cw,
                                 const float* __restrict__ cb,
                                 float* __restrict__ u,
                                 bf16_t* __restrict__ ubf) {
  const int total = M_ROWS * D_INNER;
  for (int idx = blockIdx.x * blockDim.x + threadIdx.x; idx < total;
       idx += gridDim.x * blockDim.x) {
    const int d = idx & (D_INNER - 1);
    const int l = (idx >> 11) & (L_SZ - 1);
    const int b = idx >> 22;
    float acc = cb[d];
    const float* w = cw + d * D_CONV;
#pragma unroll
    for (int j = 0; j < D_CONV; ++j) {
      const int ls = l - (D_CONV - 1) + j;
      if (ls >= 0)
        acc += w[j] * xz[(size_t)(b * L_SZ + ls) * N_XZ + d];  // x_p cols 0..2047
    }
    const float uu = acc / (1.f + __expf(-acc));  // SiLU
    u[idx]   = uu;
    ubf[idx] = (bf16_t)uu;
  }
}

// ---------------------------------------------------------------------------
// Selective scan + fused gate: per-thread channel, h[16] in registers,
// shared B/C staged through LDS in 64-step chunks.
// y_bf16 = (scan + u*D) * silu(z)
// ---------------------------------------------------------------------------
#define LCHUNK 64
__global__ __launch_bounds__(256)
void scan_gate_kernel(const float* __restrict__ dbl,   // [M][N_XPP]
                      const float* __restrict__ dt,    // [M][2048]
                      const float* __restrict__ u,     // [M][2048]
                      const float* __restrict__ xz,    // [M][4096] (z at 2048+)
                      const float* __restrict__ A_log, // [2048][16]
                      const float* __restrict__ Dp,    // [2048]
                      bf16_t* __restrict__ ybf) {      // [M][2048]
  const int b = blockIdx.x >> 3;                          // 8 blocks per batch
  const int d = ((blockIdx.x & 7) << 8) + threadIdx.x;    // channel id

  float Ac[D_STATE], h[D_STATE];
#pragma unroll
  for (int s = 0; s < D_STATE; ++s) {
    Ac[s] = -__expf(A_log[d * D_STATE + s]);
    h[s]  = 0.f;
  }
  const float Dd = Dp[d];

  __shared__ float Bsm[LCHUNK][D_STATE];
  __shared__ float Csm[LCHUNK][D_STATE];

  for (int l0 = 0; l0 < L_SZ; l0 += LCHUNK) {
    __syncthreads();
    for (int i = threadIdx.x; i < LCHUNK * D_STATE; i += 256) {
      const int ll = i >> 4, s = i & 15;
      const size_t ro = (size_t)(b * L_SZ + l0 + ll) * N_XPP;
      Bsm[ll][s] = dbl[ro + DT_RANK + s];
      Csm[ll][s] = dbl[ro + DT_RANK + D_STATE + s];
    }
    __syncthreads();
    for (int ll = 0; ll < LCHUNK; ++ll) {
      const size_t row = (size_t)(b * L_SZ + l0 + ll);
      const float dtv = dt[row * D_INNER + d];
      const float uv  = u[row * D_INNER + d];
      const float du  = dtv * uv;
      float y = 0.f;
#pragma unroll
      for (int s = 0; s < D_STATE; ++s) {
        h[s] = __expf(dtv * Ac[s]) * h[s] + du * Bsm[ll][s];
        y += h[s] * Csm[ll][s];
      }
      const float zv = xz[row * N_XZ + D_INNER + d];
      const float g  = zv / (1.f + __expf(-zv));  // silu(z)
      ybf[row * D_INNER + d] = (bf16_t)((y + uv * Dd) * g);
    }
  }
}

// ---------------------------------------------------------------------------
// Host side
// ---------------------------------------------------------------------------
extern "C" void kernel_launch(void* const* d_in, const int* in_sizes, int n_in,
                              void* d_out, int out_size, void* d_ws, size_t ws_size,
                              hipStream_t stream) {
  const float* x       = (const float*)d_in[0];
  const float* ln_g    = (const float*)d_in[1];
  const float* ln_b    = (const float*)d_in[2];
  const float* W_in    = (const float*)d_in[3];
  const float* conv_w  = (const float*)d_in[4];
  const float* conv_b  = (const float*)d_in[5];
  const float* W_xproj = (const float*)d_in[6];
  const float* W_dt    = (const float*)d_in[7];
  const float* b_dt    = (const float*)d_in[8];
  const float* A_log   = (const float*)d_in[9];
  const float* Dv      = (const float*)d_in[10];
  const float* W_out   = (const float*)d_in[11];
  float* out = (float*)d_out;

  // workspace layout (all sizes multiples of 256 B)
  char* ws = (char*)d_ws;
  size_t off = 0;
  auto take = [&](size_t bytes) { void* p = ws + off; off += (bytes + 255) & ~(size_t)255; return p; };
  bf16_t* xn_bf   = (bf16_t*)take((size_t)M_ROWS * D_MODEL * 2);
  bf16_t* Win_bf  = (bf16_t*)take((size_t)D_MODEL * N_XZ * 2);
  bf16_t* Wxp_bf  = (bf16_t*)take((size_t)D_INNER * N_XPP * 2);   // padded
  bf16_t* Wdt_bf  = (bf16_t*)take((size_t)DT_RANK * D_INNER * 2);
  bf16_t* Wout_bf = (bf16_t*)take((size_t)D_INNER * D_MODEL * 2);
  float*  xz_f    = (float*) take((size_t)M_ROWS * N_XZ * 4);
  float*  u_f     = (float*) take((size_t)M_ROWS * D_INNER * 4);
  bf16_t* u_bf    = (bf16_t*)take((size_t)M_ROWS * D_INNER * 2);
  float*  dbl_f   = (float*) take((size_t)M_ROWS * N_XPP * 4);    // padded
  bf16_t* dtin_bf = (bf16_t*)take((size_t)M_ROWS * DT_RANK * 2);
  float*  dt_f    = (float*) take((size_t)M_ROWS * D_INNER * 4);
  bf16_t* y_bf    = (bf16_t*)take((size_t)M_ROWS * D_INNER * 2);

  // 1) weights -> bf16 (W_xproj zero-padded to 256 cols)
  cvt_f32_bf16_kernel<<<4096, 256, 0, stream>>>(W_in, Win_bf, D_MODEL * N_XZ);
  cvt_pad_xproj_kernel<<<1024, 256, 0, stream>>>(W_xproj, Wxp_bf);
  cvt_f32_bf16_kernel<<<1024, 256, 0, stream>>>(W_dt, Wdt_bf, DT_RANK * D_INNER);
  cvt_f32_bf16_kernel<<<2048, 256, 0, stream>>>(W_out, Wout_bf, D_INNER * D_MODEL);

  // 2) layernorm -> bf16 activations
  layernorm_kernel<<<M_ROWS, 256, 0, stream>>>(x, ln_g, ln_b, xn_bf);

  // 3) xz = xn @ W_in               (4096 x 4096 x 1024)
  wmma_gemm_kernel<0><<<dim3(N_XZ / BN, M_ROWS / BM), 256, SMEM_BYTES, stream>>>(
      xn_bf, Win_bf, xz_f, nullptr, nullptr, nullptr,
      M_ROWS, N_XZ, D_MODEL, 0);

  // 4) depthwise conv + SiLU -> u
  conv_silu_kernel<<<16384, 256, 0, stream>>>(xz_f, conv_w, conv_b, u_f, u_bf);

  // 5) dbl = u @ W_xproj_pad        (4096 x 256 x 2048); bf16 copy of dt-rank cols
  wmma_gemm_kernel<0><<<dim3(N_XPP / BN, M_ROWS / BM), 256, SMEM_BYTES, stream>>>(
      u_bf, Wxp_bf, dbl_f, nullptr, nullptr, dtin_bf,
      M_ROWS, N_XPP, D_INNER, DT_RANK);

  // 6) dt = softplus(dtin @ W_dt + b_dt)   (4096 x 2048 x 128)
  wmma_gemm_kernel<1><<<dim3(D_INNER / BN, M_ROWS / BM), 256, SMEM_BYTES, stream>>>(
      dtin_bf, Wdt_bf, dt_f, b_dt, nullptr, nullptr,
      M_ROWS, D_INNER, DT_RANK, 0);

  // 7) selective scan + gate -> y_bf16
  scan_gate_kernel<<<B_SZ * (D_INNER / 256), 256, 0, stream>>>(
      dbl_f, dt_f, u_f, xz_f, A_log, Dv, y_bf);

  // 8) out = y @ W_out + residual   (4096 x 1024 x 2048)
  wmma_gemm_kernel<0><<<dim3(D_MODEL / BN, M_ROWS / BM), 256, SMEM_BYTES, stream>>>(
      y_bf, Wout_bf, out, nullptr, x, nullptr,
      M_ROWS, D_MODEL, D_INNER, 0);
}
